// DPCABlock_41016937676853
// MI455X (gfx1250) — matrix operand order repaired and checked by reading, not compiled, not run on hardware
//
#include <hip/hip_runtime.h>
#include <hip/hip_bf16.h>
#include <cstdint>

// ---------------------------------------------------------------------------
// CDNA5 (gfx1250) implementation of the DPCA block.
// All heavy math runs through v_wmma_f32_16x16x32_f16 (wave32 WMMA).
// ---------------------------------------------------------------------------

typedef __attribute__((ext_vector_type(16))) _Float16 v16h;
typedef __attribute__((ext_vector_type(8)))  _Float16 v8h;
typedef __attribute__((ext_vector_type(8)))  float    v8f;

#define WMMA_F16(a, b, c) \
  __builtin_amdgcn_wmma_f32_16x16x32_f16(false, (a), false, (b), (short)0, (c), false, false)

// ---- constants of this problem ----
#define BATCH   8
#define CDIM    256
#define HWDIM   4096      // 64*64
#define NTOT    32768     // BATCH*HWDIM
#define INNER   512
#define BHEADS  64        // BATCH*HEADS
#define DHEAD   64

// ---------------------------------------------------------------------------
// Fragment loaders (16-bit A 16x32 / B 32x16, contiguous k order: k = 16*(lane/16) + s)
// ---------------------------------------------------------------------------
// A fragment: row = per-lane row pointer (m row), kk = k-chunk base.
// lanes 0-15: halves 0..15 = k 0..15
// lanes 16-31: halves 0..15 = k 16..31
__device__ __forceinline__ v16h load_frag_a(const _Float16* __restrict__ row,
                                            int kk, int lane) {
  int o = (lane < 16) ? 0 : 16;
  union { v16h v; v8h h[2]; } u;
  u.h[0] = *(const v8h*)(row + kk + o);
  u.h[1] = *(const v8h*)(row + kk + 8 + o);
  return u.v;
}

// B fragment from an [n][k]-contiguous panel: base points at (n=0,k=0) of a
// 32-wide k chunk; stride = halves between consecutive n rows.
// lane n = lane%16; lanes 0-15 cover k 0..15, lanes 16-31 cover k 16..31.
__device__ __forceinline__ v16h load_frag_b(const _Float16* __restrict__ base,
                                            int stride, int lane) {
  const _Float16* p = base + (size_t)(lane & 15) * stride + ((lane < 16) ? 0 : 16);
  union { v16h v; v8h h[2]; } u;
  u.h[0] = *(const v8h*)(p);
  u.h[1] = *(const v8h*)(p + 8);
  return u.v;
}

// ---------------------------------------------------------------------------
// f32 -> f16 elementwise convert (weights)
// ---------------------------------------------------------------------------
__global__ void f32_to_f16_kernel(const float* __restrict__ x,
                                  _Float16* __restrict__ y, int n) {
  int i = blockIdx.x * blockDim.x + threadIdx.x;
  if (i < n) y[i] = (_Float16)x[i];
}

// ---------------------------------------------------------------------------
// Channel layernorm over C=256 per (b,hw) position.
// Input x: [B][C][HW] f32. Output: [n][C] f16  (n = b*HW + hw)  -- this is
// exactly the "B^T" (n-major, contiguous-k) layout the WMMA GEMM consumes.
// ---------------------------------------------------------------------------
__global__ __launch_bounds__(256)
void chan_ln_kernel(const float* __restrict__ x, const float* __restrict__ g,
                    const float* __restrict__ bb, _Float16* __restrict__ out) {
  int n  = blockIdx.x * blockDim.x + threadIdx.x;   // 0..NTOT-1
  int b  = n >> 12;
  int hw = n & (HWDIM - 1);
  const float* xp = x + (size_t)b * CDIM * HWDIM + hw;
  float s = 0.f, s2 = 0.f;
  for (int c = 0; c < CDIM; ++c) {
    float v = xp[(size_t)c * HWDIM];
    s += v; s2 += v * v;
  }
  float mean = s * (1.f / CDIM);
  float var  = s2 * (1.f / CDIM) - mean * mean;
  float rstd = rsqrtf(var + 1e-5f);
  _Float16* op = out + (size_t)n * CDIM;
  for (int c = 0; c < CDIM; ++c) {
    float v = (xp[(size_t)c * HWDIM] - mean) * rstd * g[c] + bb[c];
    op[c] = (_Float16)v;
  }
}

// ---------------------------------------------------------------------------
// WMMA GEMM:  D(M x N) = A(M x K, f16 row-major) * X(K x N)
// X supplied as Bt = X^T, layout [n][k] f16 (contiguous k per column).
// Block: 256 threads = 8 waves; block tile 256(M) x 64(N); each wave owns a
// 32x64 tile (2 A fragments x 4 B fragments -> 8 WMMAs per 32-k step, each
// B fragment reused twice). B staged in LDS as 64-wide k panels (one barrier
// pair per 64 k).
// mode 0: kv projection  (M=1024): rows <512 -> k buffer, >=512 -> v buffer,
//         stored f16 as [bh][hw][d]
// mode 1: q projection   (M=512):  f16 [bh][hw][d]
// mode 2: out projection (M=256):  f32 [n][256]
// ---------------------------------------------------------------------------
__global__ __launch_bounds__(256)
void gemm_wmma_kernel(const _Float16* __restrict__ A,
                      const _Float16* __restrict__ Bt,
                      int K, int mode,
                      _Float16* __restrict__ out_a,   // k (mode0) / q (mode1)
                      _Float16* __restrict__ out_b,   // v (mode0)
                      float* __restrict__ out_f) {    // mode2
  __shared__ __align__(16) _Float16 ldsB[64 * 64];    // [n][k] 64-wide panel, 8 KB

  int tid  = threadIdx.x;
  int lane = tid & 31;
  int wave = tid >> 5;
  int n0    = blockIdx.x * 64;
  int mbase = blockIdx.y * 256 + wave * 32;

  const _Float16* Arow0 = A + (size_t)(mbase +      (lane & 15)) * K;
  const _Float16* Arow1 = A + (size_t)(mbase + 16 + (lane & 15)) * K;

  v8f acc[2][4] = {};
  int sn = tid >> 2;            // 0..63 : n row staged by this thread
  int sc = (tid & 3) * 8;       // 0/8/16/24 : k sub-chunk (8 halves = 16 B)
  const _Float16* stage_src = Bt + (size_t)(n0 + sn) * K + sc;
  _Float16* stage_dst = ldsB + sn * 64 + sc;

  for (int kk = 0; kk < K; kk += 64) {
    __syncthreads();
    // stage B panel: 64 n x 64 k halves (8 KB)
    *(v8h*)(stage_dst)      = *(const v8h*)(stage_src + kk);
    *(v8h*)(stage_dst + 32) = *(const v8h*)(stage_src + kk + 32);
    if (kk + 64 < K) {
      __builtin_prefetch(stage_src + kk + 64, 0, 1);
      __builtin_prefetch(stage_src + kk + 96, 0, 1);
    }
    __syncthreads();

#pragma unroll
    for (int s = 0; s < 2; ++s) {
      v16h a0 = load_frag_a(Arow0, kk + s * 32, lane);
      v16h a1 = load_frag_a(Arow1, kk + s * 32, lane);
      v16h b0 = load_frag_b(ldsB +  0 * 64 + s * 32, 64, lane);
      v16h b1 = load_frag_b(ldsB + 16 * 64 + s * 32, 64, lane);
      v16h b2 = load_frag_b(ldsB + 32 * 64 + s * 32, 64, lane);
      v16h b3 = load_frag_b(ldsB + 48 * 64 + s * 32, 64, lane);
      acc[0][0] = WMMA_F16(a0, b0, acc[0][0]);
      acc[0][1] = WMMA_F16(a0, b1, acc[0][1]);
      acc[0][2] = WMMA_F16(a0, b2, acc[0][2]);
      acc[0][3] = WMMA_F16(a0, b3, acc[0][3]);
      acc[1][0] = WMMA_F16(a1, b0, acc[1][0]);
      acc[1][1] = WMMA_F16(a1, b1, acc[1][1]);
      acc[1][2] = WMMA_F16(a1, b2, acc[1][2]);
      acc[1][3] = WMMA_F16(a1, b3, acc[1][3]);
    }
  }

  // ---- epilogue ----
#pragma unroll
  for (int sub = 0; sub < 2; ++sub) {
    int mrow0 = mbase + sub * 16 + ((lane < 16) ? 0 : 8);  // 8 rows per lane
    if (mode == 2) {
#pragma unroll
      for (int tn = 0; tn < 4; ++tn) {
        int ncol = n0 + tn * 16 + (lane & 15);
        float* p = out_f + (size_t)ncol * 256 + mrow0;
        union { v8f v; float4 f[2]; } u; u.v = acc[sub][tn];
        *(float4*)p       = u.f[0];
        *(float4*)(p + 4) = u.f[1];
      }
    } else {
      int ic = mrow0;
      _Float16* dst = out_a;
      if (mode == 0 && ic >= 512) { dst = out_b; ic -= 512; }  // uniform/subtile
      int head = ic >> 6, d = ic & 63;
#pragma unroll
      for (int tn = 0; tn < 4; ++tn) {
        int ncol = n0 + tn * 16 + (lane & 15);
        int b    = ncol >> 12;
        int hw   = ncol & (HWDIM - 1);
        v8h h;
#pragma unroll
        for (int r = 0; r < 8; ++r) h[r] = (_Float16)acc[sub][tn][r];
        *(v8h*)(dst + ((size_t)((b * 8 + head) * HWDIM + hw)) * DHEAD + d) = h;
      }
    }
  }
}

// ---------------------------------------------------------------------------
// In-place L2 norm over the contiguous 64-d head vector, per (bh,hw).
// Vectorized: 8 x b128 loads / stores.
// ---------------------------------------------------------------------------
__global__ __launch_bounds__(256)
void l2norm_kernel(_Float16* __restrict__ buf) {
  size_t n = (size_t)blockIdx.x * blockDim.x + threadIdx.x;  // 0..BH*HW-1
  v8h* p = (v8h*)(buf + n * DHEAD);
  v8h v[8];
  float s = 0.f;
#pragma unroll
  for (int i = 0; i < 8; ++i) {
    v[i] = p[i];
#pragma unroll
    for (int e = 0; e < 8; ++e) { float f = (float)v[i][e]; s += f * f; }
  }
  float inv = 1.f / fmaxf(sqrtf(s), 1e-12f);
#pragma unroll
  for (int i = 0; i < 8; ++i) {
#pragma unroll
    for (int e = 0; e < 8; ++e) v[i][e] = (_Float16)((float)v[i][e] * inv);
    p[i] = v[i];
  }
}

// ---------------------------------------------------------------------------
// Top-k row/col selection per bh head.
//   q_probe[d] = sum_hw q[hw][d] ;  s(h,w) = sum_d |k[h,w,d]| * q_probe[d]
//   score_r[h] = sum_w s ; score_c[w] = sum_h s ; top-8 each (set only --
//   softmax+sum is permutation-invariant, so index order is irrelevant).
// idx layout: [bh][0..7]=rows, [bh][8..15]=cols
// ---------------------------------------------------------------------------
__global__ __launch_bounds__(256)
void topk_kernel(const _Float16* __restrict__ qb,
                 const _Float16* __restrict__ kb, int* __restrict__ idx) {
  __shared__ float part[4][64];
  __shared__ float probe[64];
  __shared__ float sr[64];
  __shared__ float sc[64];
  int bh = blockIdx.x, t = threadIdx.x;
  int d = t & 63, sl = t >> 6;
  float acc = 0.f;
  for (int hw = sl; hw < HWDIM; hw += 4)
    acc += (float)qb[((size_t)bh * HWDIM + hw) * DHEAD + d];
  part[sl][d] = acc;
  __syncthreads();
  if (t < 64) {
    probe[t] = part[0][t] + part[1][t] + part[2][t] + part[3][t];
    sr[t] = 0.f; sc[t] = 0.f;
  }
  __syncthreads();
  for (int p = t; p < HWDIM; p += 256) {
    const _Float16* kp = kb + ((size_t)bh * HWDIM + p) * DHEAD;
    float s = 0.f;
    for (int i = 0; i < DHEAD; ++i) s += fabsf((float)kp[i]) * probe[i];
    atomicAdd(&sr[p >> 6], s);   // ds_add_f32
    atomicAdd(&sc[p & 63], s);
  }
  __syncthreads();
  if (t == 0) {
    for (int r = 0; r < 8; ++r) {
      int bi = 0; float bv = sr[0];
      for (int i = 1; i < 64; ++i) if (sr[i] > bv) { bv = sr[i]; bi = i; }
      idx[bh * 16 + r] = bi; sr[bi] = -1e30f;
    }
  }
  if (t == 32) {                 // second wave does cols in parallel
    for (int c = 0; c < 8; ++c) {
      int bi = 0; float bv = sc[0];
      for (int i = 1; i < 64; ++i) if (sc[i] > bv) { bv = sc[i]; bi = i; }
      idx[bh * 16 + 8 + c] = bi; sc[bi] = -1e30f;
    }
  }
}

// ---------------------------------------------------------------------------
// Attention: per block = one bh head, 128 queries (8 waves x 16).
// Gathers top-k K/V into LDS (K as [j][d], V transposed to [d][j]), then per
// wave: sim = Q*K^T (8 WMMAs), register softmax over 64 keys (shfl width-16
// matches the D-fragment lane layout), attn bounced through LDS to become an
// A fragment, out = attn*V (8 WMMAs). Output f16 [n][INNER].
// ---------------------------------------------------------------------------
__global__ __launch_bounds__(256)
void attn_kernel(const _Float16* __restrict__ qb,
                 const _Float16* __restrict__ kb,
                 const _Float16* __restrict__ vb,
                 const int* __restrict__ idx,
                 _Float16* __restrict__ outb) {
  __shared__ __align__(16) _Float16 ldsK[64 * 64];      // [j][d]
  __shared__ __align__(16) _Float16 ldsV[64 * 64];      // [d][j]
  __shared__ __align__(16) _Float16 ldsA[8 * 16 * 64];  // per-wave attn [m][j]

  int bh = blockIdx.x >> 5;
  int qg = blockIdx.x & 31;
  int t = threadIdx.x, lane = t & 31, wave = t >> 5;

  const int* ih = idx + bh * 16;
  const int* iw = ih + 8;
  for (int e = t; e < 4096; e += 256) {
    int j = e >> 6, d = e & 63;
    int pos = ih[j >> 3] * 64 + iw[j & 7];
    size_t src = ((size_t)bh * HWDIM + pos) * DHEAD + d;
    ldsK[j * 64 + d] = kb[src];
    ldsV[d * 64 + j] = vb[src];
  }
  __syncthreads();

  int q0 = qg * 128 + wave * 16;
  const _Float16* qrow = qb + ((size_t)bh * HWDIM + q0 + (lane & 15)) * DHEAD;
  v16h aq0 = load_frag_a(qrow, 0, lane);
  v16h aq1 = load_frag_a(qrow, 32, lane);

  v8f s[4] = {};
#pragma unroll
  for (int tn = 0; tn < 4; ++tn) {
    v16h b0 = load_frag_b(ldsK + tn * 16 * 64, 64, lane);
    v16h b1 = load_frag_b(ldsK + tn * 16 * 64 + 32, 64, lane);
    s[tn] = WMMA_F16(aq0, b0, s[tn]);
    s[tn] = WMMA_F16(aq1, b1, s[tn]);
  }

  // softmax over 64 keys per query row m (= vgpr index + lane-half)
#pragma unroll
  for (int r = 0; r < 8; ++r) {
    float mx = fmaxf(fmaxf(s[0][r], s[1][r]), fmaxf(s[2][r], s[3][r]));
#pragma unroll
    for (int o = 8; o >= 1; o >>= 1) mx = fmaxf(mx, __shfl_xor(mx, o, 16));
    float sum = 0.f;
#pragma unroll
    for (int tn = 0; tn < 4; ++tn) {
      float ev = __expf(s[tn][r] - mx);
      s[tn][r] = ev; sum += ev;
    }
#pragma unroll
    for (int o = 8; o >= 1; o >>= 1) sum += __shfl_xor(sum, o, 16);
    float inv = 1.f / sum;
#pragma unroll
    for (int tn = 0; tn < 4; ++tn) s[tn][r] *= inv;
  }

  // write attn to LDS in [m][j] so it can be reloaded as an A fragment
  _Float16* ap = ldsA + wave * 1024;
  int mlo = (lane < 16) ? 0 : 8;
  int nlo = lane & 15;
#pragma unroll
  for (int tn = 0; tn < 4; ++tn)
#pragma unroll
    for (int r = 0; r < 8; ++r)
      ap[(r + mlo) * 64 + tn * 16 + nlo] = (_Float16)s[tn][r];
  __syncthreads();

  const _Float16* arow = ap + (size_t)(lane & 15) * 64;
  v16h aa0 = load_frag_a(arow, 0, lane);
  v16h aa1 = load_frag_a(arow, 32, lane);
  v8f o[4] = {};
#pragma unroll
  for (int tn = 0; tn < 4; ++tn) {
    v16h b0 = load_frag_b(ldsV + tn * 16 * 64, 64, lane);
    v16h b1 = load_frag_b(ldsV + tn * 16 * 64 + 32, 64, lane);
    o[tn] = WMMA_F16(aa0, b0, o[tn]);
    o[tn] = WMMA_F16(aa1, b1, o[tn]);
  }

  int b_ = bh >> 3, h_ = bh & 7;
#pragma unroll
  for (int tn = 0; tn < 4; ++tn) {
    int dcol = tn * 16 + nlo;
#pragma unroll
    for (int r = 0; r < 8; ++r) {
      int mq = q0 + r + mlo;
      outb[((size_t)b_ * HWDIM + mq) * INNER + h_ * 64 + dcol] = (_Float16)o[tn][r];
    }
  }
}

// ---------------------------------------------------------------------------
// Final: layernorm over C of gout[n][256], gamma*ln + residual -> d_out
// ---------------------------------------------------------------------------
__global__ __launch_bounds__(256)
void final_kernel(const float* __restrict__ gout, const float* __restrict__ g,
                  const float* __restrict__ bb, const float* __restrict__ gamma,
                  const float* __restrict__ qsrc, float* __restrict__ out) {
  int n  = blockIdx.x * blockDim.x + threadIdx.x;
  int b  = n >> 12;
  int hw = n & (HWDIM - 1);
  const float* p = gout + (size_t)n * CDIM;
  float s = 0.f, s2 = 0.f;
  for (int c = 0; c < CDIM; ++c) { float v = p[c]; s += v; s2 += v * v; }
  float mean = s * (1.f / CDIM);
  float rstd = rsqrtf(s2 * (1.f / CDIM) - mean * mean + 1e-5f);
  float gm = gamma[0];
  size_t base = (size_t)b * CDIM * HWDIM + hw;
  for (int c = 0; c < CDIM; ++c) {
    float v = (p[c] - mean) * rstd * g[c] + bb[c];
    out[base + (size_t)c * HWDIM] = gm * v + qsrc[base + (size_t)c * HWDIM];
  }
}

// ---------------------------------------------------------------------------
// launch
// ---------------------------------------------------------------------------
extern "C" void kernel_launch(void* const* d_in, const int* in_sizes, int n_in,
                              void* d_out, int out_size, void* d_ws, size_t ws_size,
                              hipStream_t stream) {
  (void)in_sizes; (void)n_in; (void)out_size; (void)ws_size;

  char* w = (char*)d_ws;
  size_t off = 0;
  auto alloc = [&](size_t bytes) -> void* {
    void* p = w + off;
    off = (off + bytes + 255) & ~(size_t)255;
    return p;
  };
  _Float16* wkv16 = (_Float16*)alloc((size_t)1024 * 256 * 2);
  _Float16* wq16  = (_Float16*)alloc((size_t)512 * 256 * 2);
  _Float16* wout16= (_Float16*)alloc((size_t)256 * 512 * 2);
  _Float16* ctxn  = (_Float16*)alloc((size_t)NTOT * CDIM * 2);
  _Float16* qsn   = (_Float16*)alloc((size_t)NTOT * CDIM * 2);
  _Float16* kbuf  = (_Float16*)alloc((size_t)BHEADS * HWDIM * DHEAD * 2);
  _Float16* vbuf  = (_Float16*)alloc((size_t)BHEADS * HWDIM * DHEAD * 2);
  _Float16* qbuf  = (_Float16*)alloc((size_t)BHEADS * HWDIM * DHEAD * 2);
  _Float16* attno = (_Float16*)alloc((size_t)NTOT * INNER * 2);
  float*    gout  = (float*)   alloc((size_t)NTOT * CDIM * 4);
  int*      idxb  = (int*)     alloc((size_t)BHEADS * 16 * 4);

  const float* qsrc = (const float*)d_in[0];
  const float* ctx  = (const float*)d_in[1];

  // weights -> f16
  f32_to_f16_kernel<<<(1024 * 256 + 255) / 256, 256, 0, stream>>>((const float*)d_in[8],  wkv16, 1024 * 256);
  f32_to_f16_kernel<<<(512  * 256 + 255) / 256, 256, 0, stream>>>((const float*)d_in[9],  wq16,  512 * 256);
  f32_to_f16_kernel<<<(256  * 512 + 255) / 256, 256, 0, stream>>>((const float*)d_in[10], wout16, 256 * 512);

  // channel layernorms -> n-major f16 activations
  chan_ln_kernel<<<NTOT / 256, 256, 0, stream>>>(ctx,  (const float*)d_in[2], (const float*)d_in[3], ctxn);
  chan_ln_kernel<<<NTOT / 256, 256, 0, stream>>>(qsrc, (const float*)d_in[4], (const float*)d_in[5], qsn);

  // projections (WMMA GEMMs); block tile 256(M) x 64(N)
  gemm_wmma_kernel<<<dim3(NTOT / 64, 1024 / 256), 256, 0, stream>>>(wkv16, ctxn, 256, 0, kbuf, vbuf, nullptr);
  gemm_wmma_kernel<<<dim3(NTOT / 64,  512 / 256), 256, 0, stream>>>(wq16,  qsn,  256, 1, qbuf, nullptr, nullptr);

  // l2 norm q, k (in place)
  l2norm_kernel<<<(BHEADS * HWDIM) / 256, 256, 0, stream>>>(qbuf);
  l2norm_kernel<<<(BHEADS * HWDIM) / 256, 256, 0, stream>>>(kbuf);

  // top-8 rows/cols per head
  topk_kernel<<<BHEADS, 256, 0, stream>>>(qbuf, kbuf, idxb);

  // attention (WMMA)
  attn_kernel<<<BHEADS * 32, 256, 0, stream>>>(qbuf, kbuf, vbuf, idxb, attno);

  // output projection (WMMA GEMM)
  gemm_wmma_kernel<<<dim3(NTOT / 64, 256 / 256), 256, 0, stream>>>(wout16, attno, 512, 2, nullptr, nullptr, gout);

  // final layernorm + gamma * out + residual
  final_kernel<<<NTOT / 256, 256, 0, stream>>>(gout, (const float*)d_in[6], (const float*)d_in[7],
                                               (const float*)d_in[11], qsrc, (float*)d_out);
}
